// NodeEdgeEncoder_17377437679638
// MI455X (gfx1250) — matrix operand order repaired and checked by reading, not compiled, not run on hardware
//
#include <hip/hip_runtime.h>

#define DIM 128
#define N_NODES 100000
#define N_EDGES 600000

// LDS strides in _Float16 units; +8 pad keeps 16B alignment and spreads
// lanes across distinct bank quads (row stride 528B -> bank offset 4/lane).
#define W1_STRIDE 264   // 256 + 8
#define W2_STRIDE 136   // 128 + 8
#define A_STRIDE  264
#define H_STRIDE  136
#define NWAVES 4

typedef __attribute__((ext_vector_type(4)))  _Float16 v4h;
typedef __attribute__((ext_vector_type(8)))  _Float16 v8h;
typedef __attribute__((ext_vector_type(16))) _Float16 v16h;
typedef __attribute__((ext_vector_type(4)))  float    v4f;
typedef __attribute__((ext_vector_type(8)))  float    v8f;

struct __align__(16) Smem {
    float    b1[DIM];                       // 512 B
    float    b2[DIM];                       // 512 B
    _Float16 w1[DIM * W1_STRIDE];           // 67584 B  (column-major: w1[n*W1_STRIDE + k])
    _Float16 w2[DIM * W2_STRIDE];           // 34816 B
    _Float16 a [NWAVES][16 * A_STRIDE];     // 4 x 8448 B  (row-major 16x256 input tile)
    _Float16 h [NWAVES][16 * H_STRIDE];     // 4 x 4352 B  (row-major 16x128 hidden tile)
};  // total 154624 B -> 2 workgroups per 320KB WGP

__device__ __forceinline__ void wave_lds_fence() {
    asm volatile("s_wait_dscnt 0" ::: "memory");
}

// mish(v) = v * tanh(softplus(v)).  With u = e^v and z = 1+u:
// tanh(ln z) = (z^2-1)/(z^2+1) = (u^2+2u)/(u^2+2u+2).
// One v_exp + one v_rcp instead of the libm exp/log/tanh chain.
__device__ __forceinline__ float mishf(float v) {
    float u = __expf(v);
    float w = __builtin_fmaf(u, u, u + u);          // u^2 + 2u
    float t = w * __builtin_amdgcn_rcpf(w + 2.0f);  // w / (w + 2)
    t = (v > 15.0f) ? 1.0f : t;                     // overflow guard (u^2 inf)
    return v * t;
}

__device__ __forceinline__ v16h cat8(v8h lo, v8h hi) {
    return __builtin_shufflevector(lo, hi, 0,1,2,3,4,5,6,7,8,9,10,11,12,13,14,15);
}

// One 16x16 output tile: C = A(16xK) * B(Kx16) + bias, K = 32*KTILES.
// Abase: row base + (lane>>4)*8 halves; Bbase: col base + (lane>>4)*8 halves.
template <int KTILES>
__device__ __forceinline__ v8f gemm_tile(const _Float16* Abase, const _Float16* Bbase, float bias) {
    v8f c = {bias, bias, bias, bias, bias, bias, bias, bias};
#pragma unroll
    for (int kt = 0; kt < KTILES; ++kt) {
        const _Float16* ap = Abase + kt * 32;
        const _Float16* bp = Bbase + kt * 32;
        v8h alo = *(const v8h*)(ap);        // K+0..7   (or K+8..15 for lanes 16-31)
        v8h ahi = *(const v8h*)(ap + 16);   // K+16..23 (or K+24..31)
        v8h blo = *(const v8h*)(bp);        // K+0..7   (or K+8..15)
        v8h bhi = *(const v8h*)(bp + 16);   // K+16..23 (or K+24..31)
        c = __builtin_amdgcn_wmma_f32_16x16x32_f16(false, cat8(alo, ahi),
                                                   false, cat8(blo, bhi),
                                                   (short)0, c, false, false);
    }
    return c;
}

// Shared 2-layer MLP body for one 16-row tile staged in A; writes 16x128 fp32 rows.
__device__ __forceinline__ void tile_mlp(Smem& sm, _Float16* A, _Float16* H,
                                         int lane, float* out_row0) {
    const int lane16 = lane & 15;
    const int hi     = lane >> 4;
    wave_lds_fence();
    const _Float16* Abase = A + lane16 * A_STRIDE + hi * 8;
    // layer 1: 16x256 @ 256x128 -> mish -> hidden LDS (f16)
#pragma unroll 2
    for (int nt = 0; nt < 8; ++nt) {
        const _Float16* Bb = sm.w1 + (nt * 16 + lane16) * W1_STRIDE + hi * 8;
        v8f c = gemm_tile<8>(Abase, Bb, sm.b1[nt * 16 + lane16]);
#pragma unroll
        for (int v = 0; v < 8; ++v)
            H[(hi * 8 + v) * H_STRIDE + nt * 16 + lane16] = (_Float16)mishf(c[v]);
    }
    wave_lds_fence();
    const _Float16* Hbase = H + lane16 * H_STRIDE + hi * 8;
    // layer 2: 16x128 @ 128x128 -> global f32
#pragma unroll 2
    for (int nt = 0; nt < 8; ++nt) {
        const _Float16* Bb = sm.w2 + (nt * 16 + lane16) * W2_STRIDE + hi * 8;
        v8f c = gemm_tile<4>(Hbase, Bb, sm.b2[nt * 16 + lane16]);
        float* op = out_row0 + (size_t)(hi * 8) * DIM + nt * 16 + lane16;
#pragma unroll
        for (int v = 0; v < 8; ++v) op[(size_t)v * DIM] = c[v];
    }
}

__device__ __forceinline__ void stage_weights(Smem& sm, int tid,
                                              const float* W1, const float* B1,
                                              const float* W2, const float* B2) {
    for (int i = tid; i < 2 * DIM * DIM; i += NWAVES * 32) {
        int k = i >> 7, n = i & 127;                // W1 row-major [256,128]
        sm.w1[n * W1_STRIDE + k] = (_Float16)W1[i];
    }
    for (int i = tid; i < DIM * DIM; i += NWAVES * 32) {
        int k = i >> 7, n = i & 127;
        sm.w2[n * W2_STRIDE + k] = (_Float16)W2[i];
    }
    for (int i = tid; i < DIM; i += NWAVES * 32) { sm.b1[i] = B1[i]; sm.b2[i] = B2[i]; }
    __syncthreads();
}

__device__ __forceinline__ void store_quad(_Float16* A, int m, int k, v4f v) {
    v4h hv = { (_Float16)v.x, (_Float16)v.y, (_Float16)v.z, (_Float16)v.w };
    *(v4h*)(A + m * A_STRIDE + k) = hv;             // ds_store_b64
}

// ---------------- edge kernel: e = MLP([edge_attr, x_src+x_dst]); agg += edge_attr @ dst ----
__global__ __launch_bounds__(NWAVES * 32) void edge_kernel(
    const float* __restrict__ x, const float* __restrict__ edge_attr,
    const int* __restrict__ eidx,
    const float* __restrict__ eW1, const float* __restrict__ eb1,
    const float* __restrict__ eW2, const float* __restrict__ eb2,
    float* __restrict__ e_out, float* __restrict__ agg) {
    extern __shared__ char smem_raw[];
    Smem& sm = *reinterpret_cast<Smem*>(smem_raw);
    const int tid = threadIdx.x;
    stage_weights(sm, tid, eW1, eb1, eW2, eb2);

    const int wave = tid >> 5, lane = tid & 31;
    const int* srcI = eidx;
    const int* dstI = eidx + N_EDGES;
    _Float16* A = sm.a[wave];
    _Float16* H = sm.h[wave];
    const int nTiles = N_EDGES / 16;

    for (int t = blockIdx.x * NWAVES + wave; t < nTiles; t += gridDim.x * NWAVES) {
        const int ebase = t * 16;
        // stage 16x256 input tile in float4 quanta (b128 loads, b64 LDS stores);
        // fuse segment_sum(edge_attr, dst) as hardware f32 atomics.
        for (int q = lane; q < 16 * 64; q += 32) {
            const int m = q >> 6, kq = q & 63, k = kq * 4;
            const int e = ebase + m;
            v4f v;
            if (kq < 32) {
                v = *(const v4f*)(edge_attr + (size_t)e * DIM + k);
                float* ap = agg + (size_t)dstI[e] * DIM + k;
                unsafeAtomicAdd(ap + 0, v.x);       // global_atomic_add_f32
                unsafeAtomicAdd(ap + 1, v.y);
                unsafeAtomicAdd(ap + 2, v.z);
                unsafeAtomicAdd(ap + 3, v.w);
            } else {
                v4f a = *(const v4f*)(x + (size_t)srcI[e] * DIM + (k - DIM));
                v4f b = *(const v4f*)(x + (size_t)dstI[e] * DIM + (k - DIM));
                v = a + b;
            }
            store_quad(A, m, k, v);
        }
        tile_mlp(sm, A, H, lane, e_out + (size_t)ebase * DIM);
    }
}

// ---------------- node kernel: x_out = MLP([x, agg]) ----------------
__global__ __launch_bounds__(NWAVES * 32) void node_kernel(
    const float* __restrict__ x, const float* __restrict__ agg,
    const float* __restrict__ nW1, const float* __restrict__ nb1,
    const float* __restrict__ nW2, const float* __restrict__ nb2,
    float* __restrict__ x_out) {
    extern __shared__ char smem_raw[];
    Smem& sm = *reinterpret_cast<Smem*>(smem_raw);
    const int tid = threadIdx.x;
    stage_weights(sm, tid, nW1, nb1, nW2, nb2);

    const int wave = tid >> 5, lane = tid & 31;
    _Float16* A = sm.a[wave];
    _Float16* H = sm.h[wave];
    const int nTiles = N_NODES / 16;  // 6250

    for (int t = blockIdx.x * NWAVES + wave; t < nTiles; t += gridDim.x * NWAVES) {
        const int nbase = t * 16;
        for (int q = lane; q < 16 * 64; q += 32) {
            const int m = q >> 6, kq = q & 63, k = kq * 4;
            const int n = nbase + m;
            v4f v = (kq < 32) ? *(const v4f*)(x   + (size_t)n * DIM + k)
                              : *(const v4f*)(agg + (size_t)n * DIM + (k - DIM));
            store_quad(A, m, k, v);
        }
        tile_mlp(sm, A, H, lane, x_out + (size_t)nbase * DIM);
    }
}

extern "C" void kernel_launch(void* const* d_in, const int* in_sizes, int n_in,
                              void* d_out, int out_size, void* d_ws, size_t ws_size,
                              hipStream_t stream) {
    (void)in_sizes; (void)n_in; (void)out_size; (void)ws_size;
    const float* x         = (const float*)d_in[0];
    const float* edge_attr = (const float*)d_in[1];
    const int*   eidx      = (const int*)  d_in[2];
    const float* eW1 = (const float*)d_in[3];
    const float* eb1 = (const float*)d_in[4];
    const float* eW2 = (const float*)d_in[5];
    const float* eb2 = (const float*)d_in[6];
    const float* nW1 = (const float*)d_in[7];
    const float* nb1 = (const float*)d_in[8];
    const float* nW2 = (const float*)d_in[9];
    const float* nb2 = (const float*)d_in[10];

    float* x_out = (float*)d_out;                       // [N_NODES, 128] first
    float* e_out = x_out + (size_t)N_NODES * DIM;       // then [N_EDGES, 128]
    float* agg   = (float*)d_ws;                        // 51.2 MB scratch

    const size_t smem_bytes = sizeof(Smem);             // ~151 KB dynamic LDS

    hipMemsetAsync(agg, 0, (size_t)N_NODES * DIM * sizeof(float), stream);
    edge_kernel<<<1024, NWAVES * 32, smem_bytes, stream>>>(
        x, edge_attr, eidx, eW1, eb1, eW2, eb2, e_out, agg);
    node_kernel<<<512, NWAVES * 32, smem_bytes, stream>>>(
        x, agg, nW1, nb1, nW2, nb2, x_out);
}
